// HeteroRGAT_52183852646759
// MI455X (gfx1250) — compile-verified, matrix operand order checked by reading, not run on hardware
//
#include <hip/hip_runtime.h>

// ---------------- problem constants (match reference) ----------------
#define F_IN   320          // IN_CH + TYPE_DIM
#define IN_CH  256
#define TDIM   64
#define HOUTC  512          // HEADS * OUT_CH
#define HEADS  8
#define RELS   8
#define NEG_SLOPE 0.2f

typedef __attribute__((ext_vector_type(16))) _Float16 v16h;
typedef __attribute__((ext_vector_type(8)))  _Float16 v8h;
typedef __attribute__((ext_vector_type(8)))  float    v8f;
typedef unsigned __attribute__((ext_vector_type(4)))  u32x4;
typedef int      __attribute__((ext_vector_type(4)))  i32x4;
typedef int      __attribute__((ext_vector_type(8)))  i32x8;

// Tensor Data Mover path only when the device compiler exposes the builtins.
#if defined(__HIP_DEVICE_COMPILE__) && \
    __has_builtin(__builtin_amdgcn_tensor_load_to_lds) && \
    __has_builtin(__builtin_amdgcn_s_wait_tensorcnt)
#define USE_TDM 1
#else
#define USE_TDM 0
#endif

// ---------------- helpers ----------------
__device__ __forceinline__ unsigned enc_f32(float f) {
    unsigned b = __float_as_uint(f);
    return (b & 0x80000000u) ? ~b : (b | 0x80000000u);   // order-preserving map
}
__device__ __forceinline__ float dec_f32(unsigned u) {
    unsigned b = (u & 0x80000000u) ? (u & 0x7FFFFFFFu) : ~u;
    return __uint_as_float(b);
}

// ---------------- kernels ----------------

// Build fp16 concat panel h = [x | type_emb[nt]]  -> [N, 320]
__global__ void pack_h_kernel(const float* __restrict__ x, const int* __restrict__ nt,
                              const float* __restrict__ temb, _Float16* __restrict__ h16,
                              int N) {
    size_t i = (size_t)blockIdx.x * blockDim.x + threadIdx.x;
    if (i >= (size_t)N * F_IN) return;
    int n = (int)(i / F_IN), f = (int)(i % F_IN);
    float v = (f < IN_CH) ? x[(size_t)n * IN_CH + f]
                          : temb[nt[n] * TDIM + (f - IN_CH)];
    h16[i] = (_Float16)v;
}

// weight [R, F_IN, HOUT] f32 -> w16t [R, HOUT, F_IN] f16 (column-major per relation)
__global__ void pack_w_kernel(const float* __restrict__ w, _Float16* __restrict__ w16t) {
    size_t i = (size_t)blockIdx.x * blockDim.x + threadIdx.x;
    if (i >= (size_t)RELS * F_IN * HOUTC) return;
    int r   = (int)(i / ((size_t)F_IN * HOUTC));
    int rem = (int)(i - (size_t)r * F_IN * HOUTC);
    int c = rem / F_IN, k = rem % F_IN;
    w16t[i] = (_Float16)w[((size_t)r * F_IN + k) * HOUTC + c];
}

__global__ void zero_u32_kernel(unsigned* __restrict__ p, size_t n) {
    size_t i = (size_t)blockIdx.x * blockDim.x + threadIdx.x;
    if (i < n) p[i] = 0u;
}

// xw[r] = h @ weight[r] via v_wmma_f32_16x16x32_f16.
// WG = 256 threads = 8 waves; WG tile = 64 rows x 512 cols; wave tile = 64 x 64
// (4 m-tiles x 4 n-tiles = 16 accumulators). A panel staged in LDS via TDM.
#define GROWS 64
__global__ void __launch_bounds__(256)
gemm_xw_kernel(const _Float16* __restrict__ h16, const _Float16* __restrict__ w16t,
               float* __restrict__ xw, int N) {
#if USE_TDM
    constexpr int AST = F_IN;        // TDM writes rows contiguously
#else
    constexpr int AST = F_IN + 8;    // manual path: skew to spread LDS banks
#endif
    __shared__ __align__(16) _Float16 As[GROWS * AST];
    const int r    = blockIdx.y;
    const int row0 = blockIdx.x * GROWS;
    const int tid  = threadIdx.x;
    const int wave = tid >> 5;
    const int lane = tid & 31;

#if USE_TDM
    if (wave == 0) {
        int rows = N - row0; if (rows > GROWS) rows = GROWS;
        unsigned long long ga = (unsigned long long)(h16 + (size_t)row0 * F_IN);
        unsigned lds = (unsigned)(unsigned long long)(&As[0]);
        // D# Group0: count=1 | lds_addr | global_addr[56:0] | type=2 (ISA 8.3)
        u32x4 g0 = { 1u,
                     lds,
                     (unsigned)(ga & 0xFFFFFFFFull),
                     (unsigned)(((ga >> 32) & 0x1FFFFFFull) | (2u << 30)) };
        // D# Group1: data_size=1 (2B) | tensor_dim0=F_IN | tensor_dim1=N |
        //            tile_dim0=F_IN | tile_dim1=rows | tensor_dim0_stride=F_IN (ISA 8.4)
        i32x8 g1 = { (int)(1u << 16),
                     (int)((unsigned)F_IN << 16),
                     (int)(((unsigned)N & 0xFFFFu) << 16),
                     (int)((((unsigned)N >> 16) & 0xFFFFu) | ((unsigned)F_IN << 16)),
                     (int)(unsigned)rows,
                     (int)F_IN,
                     0, 0 };
        i32x4 z4 = { 0, 0, 0, 0 };
#if __clang_major__ >= 23
        i32x8 z8 = { 0, 0, 0, 0, 0, 0, 0, 0 };
        __builtin_amdgcn_tensor_load_to_lds(g0, g1, z4, z4, z8, 0);
#else
        __builtin_amdgcn_tensor_load_to_lds(g0, g1, z4, z4, 0);
#endif
        __builtin_amdgcn_s_wait_tensorcnt(0);
    }
    __syncthreads();
#else
    {
        unsigned* s = (unsigned*)As;
        for (int i = tid; i < GROWS * (F_IN / 2); i += 256) {
            int row = i / (F_IN / 2), c = i - row * (F_IN / 2);
            int grow = row0 + row;
            unsigned v = 0u;
            if (grow < N)
                v = ((const unsigned*)h16)[(size_t)grow * (F_IN / 2) + c];
            s[row * (AST / 2) + c] = v;
        }
    }
    __syncthreads();
#endif

    const int colbase = wave * 64;
    const int kg = (lane >> 4) << 3;    // 0 for lanes 0-15, 8 for lanes 16-31
    const int m  = lane & 15;

    v8f acc[4][4] = {};   // [mtile][ntile]

    for (int k0 = 0; k0 < F_IN; k0 += 32) {
        // B fragments for this wave's 4 column tiles
        v16h b[4];
#pragma unroll
        for (int t = 0; t < 4; ++t) {
            const _Float16* bp =
                w16t + ((size_t)r * HOUTC + colbase + t * 16 + m) * F_IN + k0 + kg;
            if (k0 + 32 < F_IN) __builtin_prefetch(bp + 32, 0, 3);  // global_prefetch_b8
            const v8h lo = *(const v8h*)(bp);
            const v8h hi = *(const v8h*)(bp + 16);
#pragma unroll
            for (int i = 0; i < 8; ++i) { b[t][i] = lo[i]; b[t][i + 8] = hi[i]; }
        }
        // 4 A fragments (16 rows each) x 4 B fragments = 16 WMMAs per K step
#pragma unroll
        for (int mt = 0; mt < 4; ++mt) {
            v16h a;
            {
                const _Float16* ap = As + (mt * 16 + m) * AST + k0 + kg;
                const v8h lo = *(const v8h*)(ap);
                const v8h hi = *(const v8h*)(ap + 16);
#pragma unroll
                for (int i = 0; i < 8; ++i) { a[i] = lo[i]; a[i + 8] = hi[i]; }
            }
#pragma unroll
            for (int t = 0; t < 4; ++t) {
                acc[mt][t] = __builtin_amdgcn_wmma_f32_16x16x32_f16(
                    false, a, false, b[t], (short)0, acc[mt][t], false, false);
            }
        }
    }

    // Store D: VGPR v -> row = row0 + mt*16 + (lane/16)*8 + v ; col = colbase + t*16 + lane%16
    const int mrow = (lane >> 4) << 3;
#pragma unroll
    for (int mt = 0; mt < 4; ++mt) {
#pragma unroll
        for (int t = 0; t < 4; ++t) {
            const int col = colbase + t * 16 + m;
#pragma unroll
            for (int v = 0; v < 8; ++v) {
                int row = row0 + mt * 16 + mrow + v;
                if (row < N)
                    xw[((size_t)r * N + row) * HOUTC + col] = acc[mt][t][v];
            }
        }
    }
}

// qn/kn: wave-per-(r,node) dot products against att_q/att_k staged in LDS.
__global__ void __launch_bounds__(256)
qk_kernel(const float* __restrict__ xw, const float* __restrict__ attq,
          const float* __restrict__ attk, float* __restrict__ qn,
          float* __restrict__ kn, int N) {
    __shared__ float sq[HOUTC * HEADS];
    __shared__ float sk[HOUTC * HEADS];
    const int r = blockIdx.y;
    const int tid = threadIdx.x;
    for (int i = tid; i < HOUTC * HEADS; i += 256) {
        sq[i] = attq[(size_t)r * HOUTC * HEADS + i];
        sk[i] = attk[(size_t)r * HOUTC * HEADS + i];
    }
    __syncthreads();
    const int wave = tid >> 5, lane = tid & 31;
    const int n = blockIdx.x * 8 + wave;
    if (n >= N) return;

    const float* row = xw + ((size_t)r * N + n) * HOUTC + lane * 16;
    float q[HEADS] = {}, k[HEADS] = {};
#pragma unroll
    for (int i = 0; i < 16; ++i) {
        float v = row[i];
        int c = lane * 16 + i;
#pragma unroll
        for (int h = 0; h < HEADS; ++h) {
            q[h] += v * sq[c * HEADS + h];
            k[h] += v * sk[c * HEADS + h];
        }
    }
#pragma unroll
    for (int h = 0; h < HEADS; ++h) {
#pragma unroll
        for (int off = 16; off > 0; off >>= 1) {
            q[h] += __shfl_xor(q[h], off, 32);
            k[h] += __shfl_xor(k[h], off, 32);
        }
    }
    if (lane == 0) {
#pragma unroll
        for (int h = 0; h < HEADS; ++h) {
            qn[((size_t)r * N + n) * HEADS + h] = q[h];
            kn[((size_t)r * N + n) * HEADS + h] = k[h];
        }
    }
}

// alpha = leaky_relu(qn[rt,dst] + kn[rt,src]); segment-max via ordered-uint atomicMax.
__global__ void edge_alpha_max_kernel(const int* __restrict__ eidx, const int* __restrict__ etype,
                                      const float* __restrict__ qn, const float* __restrict__ kn,
                                      float* __restrict__ alpha, unsigned* __restrict__ amax,
                                      int N, int E) {
    size_t i = (size_t)blockIdx.x * blockDim.x + threadIdx.x;
    if (i >= (size_t)E * HEADS) return;
    int e = (int)(i >> 3), h = (int)(i & 7);
    int src = eidx[e], dst = eidx[(size_t)E + e], rt = etype[e];
    float a = qn[((size_t)rt * N + dst) * HEADS + h] + kn[((size_t)rt * N + src) * HEADS + h];
    a = (a > 0.f) ? a : NEG_SLOPE * a;
    alpha[i] = a;
    atomicMax(amax + (size_t)dst * HEADS + h, enc_f32(a));
}

// expa = exp(alpha - amax[dst]); segment-sum into denom.
__global__ void edge_exp_sum_kernel(const int* __restrict__ eidx, float* __restrict__ alpha,
                                    const unsigned* __restrict__ amax, float* __restrict__ denom,
                                    int E) {
    size_t i = (size_t)blockIdx.x * blockDim.x + threadIdx.x;
    if (i >= (size_t)E * HEADS) return;
    int e = (int)(i >> 3), h = (int)(i & 7);
    int dst = eidx[(size_t)E + e];
    float m = dec_f32(amax[(size_t)dst * HEADS + h]);
    float ex = __expf(alpha[i] - m);
    alpha[i] = ex;   // reuse buffer for expa
    atomicAdd(denom + (size_t)dst * HEADS + h, ex);
}

// out[dst] += (expa/denom) * xw[rt, src]  — wave-per-edge, lane-strided channels.
__global__ void __launch_bounds__(256)
scatter_kernel(const int* __restrict__ eidx, const int* __restrict__ etype,
               const float* __restrict__ expa, const float* __restrict__ denom,
               const float* __restrict__ xw, float* __restrict__ out, int N, int E) {
    const int wave = threadIdx.x >> 5, lane = threadIdx.x & 31;
    const int e = blockIdx.x * 8 + wave;
    if (e >= E) return;
    const int src = eidx[e], dst = eidx[(size_t)E + e], rt = etype[e];
    const float* xrow = xw + ((size_t)rt * N + src) * HOUTC;
    float* orow = out + (size_t)dst * HOUTC;
#pragma unroll
    for (int i = 0; i < 16; ++i) {
        int c = lane + i * 32;
        int h = c >> 6;
        float attn = expa[(size_t)e * HEADS + h] /
                     (denom[(size_t)dst * HEADS + h] + 1e-16f);
        atomicAdd(orow + c, attn * xrow[c]);
    }
}

// out = relu(out + bias)
__global__ void finalize_kernel(float* __restrict__ out, const float* __restrict__ bias,
                                size_t total) {
    size_t i = (size_t)blockIdx.x * blockDim.x + threadIdx.x;
    if (i >= total) return;
    float v = out[i] + bias[i & (HOUTC - 1)];
    out[i] = (v > 0.f) ? v : 0.f;
}

// ---------------- launcher ----------------
extern "C" void kernel_launch(void* const* d_in, const int* in_sizes, int n_in,
                              void* d_out, int out_size, void* d_ws, size_t ws_size,
                              hipStream_t stream) {
    (void)n_in; (void)out_size; (void)ws_size;
    const float* x     = (const float*)d_in[0];
    const int*   nt    = (const int*)  d_in[1];
    const int*   eidx  = (const int*)  d_in[2];   // [2, E]: row0 = src, row1 = dst
    const int*   etype = (const int*)  d_in[3];
    const float* temb  = (const float*)d_in[4];
    const float* weight= (const float*)d_in[5];
    const float* attq  = (const float*)d_in[6];
    const float* attk  = (const float*)d_in[7];
    const float* bias  = (const float*)d_in[8];
    float* out = (float*)d_out;

    const int N = in_sizes[1];
    const int E = in_sizes[3];

    // workspace carve (256B aligned chunks)
    size_t off = 0;
    auto carve = [&](size_t bytes) -> char* {
        char* p = (char*)d_ws + off;
        off += (bytes + 255) & ~(size_t)255;
        return p;
    };
    _Float16* h16   = (_Float16*)carve((size_t)N * F_IN * 2);
    _Float16* w16t  = (_Float16*)carve((size_t)RELS * F_IN * HOUTC * 2);
    float*    xw    = (float*)   carve((size_t)RELS * N * HOUTC * 4);
    float*    qn    = (float*)   carve((size_t)RELS * N * HEADS * 4);
    float*    kn    = (float*)   carve((size_t)RELS * N * HEADS * 4);
    unsigned* amax  = (unsigned*)carve((size_t)N * HEADS * 4);
    float*    denom = (float*)   carve((size_t)N * HEADS * 4);
    float*    alpha = (float*)   carve((size_t)E * HEADS * 4);

    const int T = 256;
    auto nb = [](size_t n, int t) { return (unsigned)((n + t - 1) / t); };

    // 1) pack inputs
    pack_h_kernel<<<nb((size_t)N * F_IN, T), T, 0, stream>>>(x, nt, temb, h16, N);
    pack_w_kernel<<<nb((size_t)RELS * F_IN * HOUTC, T), T, 0, stream>>>(weight, w16t);

    // 2) zero-init accumulators (amax encoding: 0 == "below every float")
    zero_u32_kernel<<<nb((size_t)N * HEADS, T), T, 0, stream>>>(amax, (size_t)N * HEADS);
    zero_u32_kernel<<<nb((size_t)N * HEADS, T), T, 0, stream>>>((unsigned*)denom, (size_t)N * HEADS);
    zero_u32_kernel<<<nb((size_t)N * HOUTC, T), T, 0, stream>>>((unsigned*)out, (size_t)N * HOUTC);

    // 3) WMMA GEMM: xw[r] = h @ W[r]   (TDM-staged A panel, 64x512 WG tiles)
    dim3 ggrid((unsigned)((N + GROWS - 1) / GROWS), RELS);
    gemm_xw_kernel<<<ggrid, T, 0, stream>>>(h16, w16t, xw, N);

    // 4) q/k projections
    dim3 qgrid((unsigned)((N + 7) / 8), RELS);
    qk_kernel<<<qgrid, T, 0, stream>>>(xw, attq, attk, qn, kn, N);

    // 5) softmax over incoming edges (max -> exp/sum -> normalize in scatter)
    edge_alpha_max_kernel<<<nb((size_t)E * HEADS, T), T, 0, stream>>>(eidx, etype, qn, kn,
                                                                      alpha, amax, N, E);
    edge_exp_sum_kernel<<<nb((size_t)E * HEADS, T), T, 0, stream>>>(eidx, alpha, amax, denom, E);

    // 6) weighted message scatter-add
    scatter_kernel<<<(unsigned)((E + 7) / 8), T, 0, stream>>>(eidx, etype, alpha, denom,
                                                              xw, out, N, E);

    // 7) bias + relu
    finalize_kernel<<<nb((size_t)N * HOUTC, T), T, 0, stream>>>(out, bias, (size_t)N * HOUTC);
}
